// InteractionNet_60962765800175
// MI455X (gfx1250) — matrix operand order repaired and checked by reading, not compiled, not run on hardware
//
#include <hip/hip_runtime.h>
#include <hip/hip_bf16.h>

// ---------------------------------------------------------------------------
// Problem constants (from reference)
// ---------------------------------------------------------------------------
#define BB    64
#define NVEC  4096
#define NDIM  64
#define NLNK  13
#define NW    13
#define HF    128

#define MROWS ((long)BB * NVEC)      // 262144 flattened (b, n) rows

// LDS row strides (elements, bf16). +2 halfwords => 33-dword stride, odd vs 64
// banks => conflict-free column reads across 16 rows.
#define STR_K64  66
#define STR_K128 130

typedef __attribute__((ext_vector_type(16))) __bf16 v16bf;
typedef __attribute__((ext_vector_type(8)))  float  v8f;

union FragAB {
    v16bf v;
    unsigned int u[8];
};

static __device__ __forceinline__ unsigned short f2bf(float f) {
    __bf16 h = (__bf16)f;
    union { __bf16 h; unsigned short u; } c;
    c.h = h;
    return c.u;
}

// Exact-GELU via branchless Abramowitz–Stegun 7.1.26 erf (|err| < 1.5e-7).
// ~13 straight-line VALU ops (v_rcp_f32 + v_exp_f32 + fmas), no EXEC branches.
static __device__ __forceinline__ float gelu_fast(float x) {
    const float ax = __builtin_fabsf(x) * 0.70710678118654752f;   // |x|/sqrt(2)
    const float t  = __builtin_amdgcn_rcpf(fmaf(0.3275911f, ax, 1.0f));
    float p = fmaf(1.061405429f, t, -1.453152027f);
    p = fmaf(p, t, 1.421413741f);
    p = fmaf(p, t, -0.284496736f);
    p = fmaf(p, t, 0.254829592f);
    p = p * t;
    const float e = __builtin_amdgcn_exp2f(-1.4426950408889634f * ax * ax);
    const float erf_ax = fmaf(-p, e, 1.0f);                       // erf(|x|/sqrt2)
    const float erf_x  = __builtin_copysignf(erf_ax, x);
    return 0.5f * x * (1.0f + erf_x);
}

// Fire-and-forget async global -> LDS copy (16B), tracked by ASYNCcnt.
// Generic LDS pointer truncates to the 32-bit LDS byte offset (ISA 10.2:
// LDS aperture addresses map to LDS by discarding the upper bits).
static __device__ __forceinline__ void async_load_b128(void* lds_dst, const void* gsrc) {
    const unsigned int lds_off = (unsigned int)(unsigned long long)lds_dst;
    asm volatile("global_load_async_to_lds_b128 %0, %1, off"
                 :: "v"(lds_off), "v"(gsrc) : "memory");
}

static __device__ __forceinline__ void wait_asynccnt0() {
    asm volatile("s_wait_asynccnt 0x0" ::: "memory");
}

// ---------------------------------------------------------------------------
// Fused MLP:  out[M, OUT_COLS] = GELU(X[M,64] @ W1[64,128] + b1) @ W2[128,OUT_COLS] + b2
// One block = 128 rows, 8 waves, each wave owns a 16-row strip.
// All GEMM work via v_wmma_f32_16x16x32_bf16; hidden activation lives in LDS only.
// ---------------------------------------------------------------------------
template <int OD_PAD, int OUT_COLS>
__global__ __launch_bounds__(256)
void mlp_fused(const float* __restrict__ X,
               const float* __restrict__ W1,   // [64, 128] row-major (d, h)
               const float* __restrict__ b1,   // [128]
               const float* __restrict__ W2,   // [128, OUT_COLS] row-major (h, o)
               const float* __restrict__ b2,   // [OUT_COLS]
               float* __restrict__ out)        // [M, OUT_COLS]
{
    __shared__ __attribute__((aligned(16))) unsigned short sX [128 * STR_K64];   // [row][d]
    __shared__ __attribute__((aligned(16))) unsigned short sW1[128 * STR_K64];   // transposed: [h][d]
    __shared__ __attribute__((aligned(16))) unsigned short sH [128 * STR_K128];  // [row][h]
    __shared__ __attribute__((aligned(16))) unsigned short sW2[OD_PAD * STR_K128]; // transposed: [o][h]
    __shared__ float sB1[128];
    __shared__ float sB2[OD_PAD];

    const int tid  = threadIdx.x;
    const int lane = tid & 31;
    const int wave = tid >> 5;
    const int m16  = lane & 15;      // N-index / row-index inside a 16-half
    const int hi   = lane >> 4;      // lane-half selector
    const long mBase = (long)blockIdx.x * 128;

    // ---- stage X tile: 128 rows x 64 f32 -> bf16 LDS (b128 global loads) ----
    {
        const int row = tid >> 1;
        const int cb  = (tid & 1) * 32;
        const float4* src4 = reinterpret_cast<const float4*>(X + (mBase + row) * NDIM + cb);
        unsigned short* dst = &sX[row * STR_K64 + cb];
#pragma unroll
        for (int i = 0; i < 8; ++i) {
            float4 t = src4[i];
            dst[4 * i + 0] = f2bf(t.x);
            dst[4 * i + 1] = f2bf(t.y);
            dst[4 * i + 2] = f2bf(t.z);
            dst[4 * i + 3] = f2bf(t.w);
        }
    }

    // ---- stage W1 transposed: sW1[h][d] ----
#pragma unroll
    for (int i = 0; i < (64 * 128) / 256; ++i) {
        const int g = i * 256 + tid;          // coalesced global read
        const int d = g >> 7;
        const int h = g & 127;
        sW1[h * STR_K64 + d] = f2bf(W1[g]);
    }

    // ---- stage W2 transposed + zero-padded: sW2[o][h] ----
#pragma unroll
    for (int i = 0; i < (OD_PAD * 128) / 256; ++i) {
        const int g = i * 256 + tid;
        const int o = g % OD_PAD;
        const int h = g / OD_PAD;
        const float v = (o < OUT_COLS) ? W2[h * OUT_COLS + o] : 0.0f;
        sW2[o * STR_K128 + h] = f2bf(v);
    }

    if (tid < 128) sB1[tid] = b1[tid];
    if (tid < OD_PAD) sB2[tid] = (tid < OUT_COLS) ? b2[tid] : 0.0f;

    __syncthreads();

    const int rowB = wave * 16;   // this wave's 16-row strip (local)

    // ---- stage 1: H = GELU(X @ W1 + b1), K=64 split into two K=32 steps ----
    // A-fragment layout (16-bit A 16x32, ISA 7.12.2):
    //   lane half hi, VGPR v holds K pair: k = 2v + (v>=4 ? 8 : 0) + 8*hi
    FragAB a0, a1;
#pragma unroll
    for (int v = 0; v < 8; ++v) {
        const int k = 2 * v + ((v >= 4) ? 8 : 0) + 8 * hi;
        const unsigned short* p = &sX[(rowB + m16) * STR_K64 + k];
        a0.u[v] = *(const unsigned int*)(p);        // K base 0
        a1.u[v] = *(const unsigned int*)(p + 32);   // K base 32
    }

#pragma unroll
    for (int j = 0; j < 8; ++j) {                   // 8 x 16 hidden columns
        // B-fragment (16-bit B 32x16): same K order as A: VGPR v packs k = 2v + (v>=4 ? 8 : 0) + 8*hi, +1
        FragAB bA, bB;
#pragma unroll
        for (int v = 0; v < 8; ++v) {
            const int kk = 2 * v + ((v >= 4) ? 8 : 0) + 8 * hi;
            const unsigned short* p = &sW1[(j * 16 + m16) * STR_K64 + kk];
            bA.u[v] = *(const unsigned int*)(p);
            bB.u[v] = *(const unsigned int*)(p + 32);
        }
        v8f acc = {0.f, 0.f, 0.f, 0.f, 0.f, 0.f, 0.f, 0.f};
        acc = __builtin_amdgcn_wmma_f32_16x16x32_bf16(false, a0.v, false, bA.v,
                                                      (short)0, acc, false, false);
        acc = __builtin_amdgcn_wmma_f32_16x16x32_bf16(false, a1.v, false, bB.v,
                                                      (short)0, acc, false, false);
        const float bias = sB1[j * 16 + m16];
        // C/D layout: VGPR r -> M = r + 8*hi, N = m16
#pragma unroll
        for (int r = 0; r < 8; ++r) {
            const float g = gelu_fast(acc[r] + bias);
            sH[(rowB + r + 8 * hi) * STR_K128 + (j * 16 + m16)] = f2bf(g);
        }
    }

    __syncthreads();

    // ---- stage 2: out = H @ W2 + b2, K=128 as four K=32 steps ----
    FragAB ah[4];
#pragma unroll
    for (int s = 0; s < 4; ++s) {
#pragma unroll
        for (int v = 0; v < 8; ++v) {
            const int k = 32 * s + 2 * v + ((v >= 4) ? 8 : 0) + 8 * hi;
            ah[s].u[v] = *(const unsigned int*)&sH[(rowB + m16) * STR_K128 + k];
        }
    }

#pragma unroll
    for (int jo = 0; jo < OD_PAD / 16; ++jo) {
        v8f acc = {0.f, 0.f, 0.f, 0.f, 0.f, 0.f, 0.f, 0.f};
#pragma unroll
        for (int s = 0; s < 4; ++s) {
            FragAB bb;
#pragma unroll
            for (int v = 0; v < 8; ++v) {
                const int kk = 32 * s + 2 * v + ((v >= 4) ? 8 : 0) + 8 * hi;
                bb.u[v] = *(const unsigned int*)&sW2[(jo * 16 + m16) * STR_K128 + kk];
            }
            acc = __builtin_amdgcn_wmma_f32_16x16x32_bf16(false, ah[s].v, false, bb.v,
                                                          (short)0, acc, false, false);
        }
        const int n = jo * 16 + m16;
        if (n < OUT_COLS) {
            const float bias = sB2[n];
#pragma unroll
            for (int r = 0; r < 8; ++r) {
                const long m = mBase + rowB + r + 8 * hi;
                out[m * OUT_COLS + n] = acc[r] + bias;
            }
        }
    }
}

// ---------------------------------------------------------------------------
// Butterfly sparse multiply: Zout[b,v,:] = sum_l F[b,v,l] * Zin[b,(v+2^(l-1))%N,:]
//
// For a 16-row block, the union of the 13 gather targets is compact:
//   rows [vBase, vBase+32)            covers offsets {0,1,2,4,8,16}
//   seven 16-row strips at 32<<i      covers offsets {32,...,2048}
// = 144 rows x 64 f32 = 36 KB. Stage them with fire-and-forget
// GLOBAL_LOAD_ASYNC_TO_LDS_B128 (ASYNCcnt), drain with s_wait_asynccnt 0,
// then run the 13-term FMA reduction entirely out of LDS.
// ---------------------------------------------------------------------------
#define SPZ_ROWS 144   // 32 + 7*16

__global__ __launch_bounds__(256)
void sparse_mult(const float* __restrict__ F,     // [B, N, 13]
                 const float* __restrict__ Zin,   // [B, N, 64]
                 float* __restrict__ Zout)        // [B, N, 64]
{
    __shared__ __attribute__((aligned(16))) float sZ[SPZ_ROWS * NDIM];  // 36 KB
    __shared__ float sF[16 * NLNK];

    const int tid   = threadIdx.x;
    const int b     = blockIdx.x >> 8;           // N/16 = 256 row-blocks per batch
    const int vBase = (blockIdx.x & 255) << 4;
    const int r     = tid >> 4;                  // row within block (0..15)
    const int d4    = tid & 15;                  // float4 column (0..15)
    const int v     = vBase + r;

    const float* Zb = Zin + (size_t)b * NVEC * NDIM;

    // ---- async-stage the 144-row gather union into LDS (9 b128 per thread) ----
#pragma unroll
    for (int i = 0; i < (SPZ_ROWS * 16) / 256; ++i) {
        const int c  = i * 256 + tid;            // float4 chunk id
        const int rl = c >> 4;                   // LDS row (0..143)
        const int c4 = c & 15;                   // float4 within row
        int goff;                                // row offset from vBase
        if (rl < 32) {
            goff = rl;
        } else {
            const int s = (rl - 32) >> 4;        // strip 0..6 -> offset 32<<s
            goff = (32 << s) + ((rl - 32) & 15);
        }
        const int srow = (vBase + goff) & (NVEC - 1);
        async_load_b128(&sZ[rl * NDIM + c4 * 4],
                        Zb + (size_t)srow * NDIM + c4 * 4);
    }

    // Stage the 16 rows of link weights (coalesced 208 floats).
    if (tid < 16 * NLNK)
        sF[tid] = F[((size_t)b * NVEC + vBase) * NLNK + tid];

    wait_asynccnt0();        // this wave's async copies landed in LDS
    __syncthreads();         // all waves' copies visible

    float f[NLNK];
#pragma unroll
    for (int l = 0; l < NLNK; ++l) f[l] = sF[r * NLNK + l];

    constexpr int OFF[NLNK] = {0, 1, 2, 4, 8, 16, 32, 64, 128, 256, 512, 1024, 2048};

    float ax = 0.f, ay = 0.f, az = 0.f, aw = 0.f;
#pragma unroll
    for (int l = 0; l < NLNK; ++l) {
        // LDS row holding Zin[b, (v+OFF[l]) % N, :]
        const int rl = (l < 6) ? (r + OFF[l]) : (32 + 16 * (l - 6) + r);
        const float4 z = *(const float4*)&sZ[rl * NDIM + d4 * 4];
        ax = fmaf(f[l], z.x, ax);
        ay = fmaf(f[l], z.y, ay);
        az = fmaf(f[l], z.z, az);
        aw = fmaf(f[l], z.w, aw);
    }
    float4 o; o.x = ax; o.y = ay; o.z = az; o.w = aw;
    *(float4*)(Zout + ((size_t)b * NVEC + v) * NDIM + d4 * 4) = o;
}

// ---------------------------------------------------------------------------
// Launch: VNet -> (FNet_k -> butterfly) x 13, Z ping-pongs through workspace,
// final sweep lands in d_out.
// ---------------------------------------------------------------------------
extern "C" void kernel_launch(void* const* d_in, const int* in_sizes, int n_in,
                              void* d_out, int out_size, void* d_ws, size_t ws_size,
                              hipStream_t stream)
{
    (void)in_sizes; (void)n_in; (void)out_size; (void)ws_size;

    const float* inputs = (const float*)d_in[0];
    const float* fW1 = (const float*)d_in[1];
    const float* fb1 = (const float*)d_in[2];
    const float* fW2 = (const float*)d_in[3];
    const float* fb2 = (const float*)d_in[4];
    const float* vW1 = (const float*)d_in[5];
    const float* vb1 = (const float*)d_in[6];
    const float* vW2 = (const float*)d_in[7];
    const float* vb2 = (const float*)d_in[8];
    float* out = (float*)d_out;

    float* Za   = (float*)d_ws;                 // [M, 64]
    float* Zb   = Za + MROWS * NDIM;            // [M, 64]
    float* Fbuf = Zb + MROWS * NDIM;            // [M, 13]

    const int gemmBlocks   = (int)(MROWS / 128);
    const int sparseBlocks = BB * (NVEC / 16);

    // VNet init: Z = MLP_v(inputs)
    mlp_fused<64, 64><<<gemmBlocks, 256, 0, stream>>>(inputs, vW1, vb1, vW2, vb2, Za);

    float* zin = Za;
    float* zout = Zb;
    for (int k = 0; k < NW; ++k) {
        mlp_fused<16, 13><<<gemmBlocks, 256, 0, stream>>>(
            inputs,
            fW1 + (size_t)k * NDIM * HF,
            fb1 + (size_t)k * HF,
            fW2 + (size_t)k * HF * NLNK,
            fb2 + (size_t)k * NLNK,
            Fbuf);

        float* dst = (k == NW - 1) ? out : zout;
        sparse_mult<<<sparseBlocks, 256, 0, stream>>>(Fbuf, zin, dst);

        float* t = zin; zin = zout; zout = t;
    }
}
